// ChannelRowAttention_42331197669731
// MI455X (gfx1250) — matrix operand order repaired and checked by reading, not compiled, not run on hardware
//
#include <hip/hip_runtime.h>
#include <hip/hip_bf16.h>

typedef __attribute__((ext_vector_type(16))) __bf16 v16bf;
typedef __attribute__((ext_vector_type(8)))  __bf16 v8bf;
typedef __attribute__((ext_vector_type(2)))  __bf16 v2bf;
typedef __attribute__((ext_vector_type(8)))  float  v8f;
typedef __attribute__((ext_vector_type(4)))  float  v4f;
typedef __attribute__((ext_vector_type(4)))  unsigned int v4u;
typedef __attribute__((ext_vector_type(8)))  int    v8i;
typedef __attribute__((ext_vector_type(4)))  int    v4i;

#define IN_C 256
#define QK   64
#define H    128
#define W    128
#define NBATCH 8
#define HW   (H * W)

// ---------------- WMMA fragment helpers (16x16x32 bf16, wave32) ----------------
// A (16x32, MxK) from row-major LDS: lane m = lane&15; K chunks [k0,k0+8) and
// [k0+16,k0+24), k0 = 8*(lane>>4)  -> two ds_load_b128
__device__ inline v16bf load_a_lds(const __bf16* A, int lda, int lane) {
    int m  = lane & 15;
    int k0 = (lane >> 4) << 3;
    const __bf16* p = A + m * lda + k0;
    v8bf lo = *(const v8bf*)(p);
    v8bf hi = *(const v8bf*)(p + 16);
    v16bf a;
#pragma unroll
    for (int i = 0; i < 8; ++i) { a[i] = lo[i]; a[8 + i] = hi[i]; }
    return a;
}
// A fragment sourced from a row-major f32 global matrix (converted to bf16)
__device__ inline v16bf load_a_glob_f32(const float* A, int lda, int lane) {
    int m  = lane & 15;
    int k0 = (lane >> 4) << 3;
    const float* p = A + m * lda + k0;
    v16bf a;
#pragma unroll
    for (int i = 0; i < 8; ++i) { a[i] = (__bf16)p[i]; a[8 + i] = (__bf16)p[16 + i]; }
    return a;
}
// B (32x16, KxN) fragment loaded from the TRANSPOSE of B stored row-major in LDS:
// per lane (col n = lane&15, K chunks [k0,k0+8) and [k0+16,k0+24), k0 = 8*(lane>>4))
__device__ inline v16bf load_b_ldsT(const __bf16* Bt, int ldt, int lane) {
    int n  = lane & 15;
    int k0 = (lane >> 4) << 3;
    const __bf16* p = Bt + n * ldt + k0;
    v8bf lo = *(const v8bf*)(p);
    v8bf hi = *(const v8bf*)(p + 16);
    v16bf b;
#pragma unroll
    for (int i = 0; i < 8; ++i) { b[i] = lo[i]; b[8 + i] = hi[i]; }
    return b;
}
// B[k][n] = Wt[n][k]  (Wt row-major NxK f32 in global: transposed weight as B)
__device__ inline v16bf load_bT_glob_f32(const float* Wt, int ldw, int lane) {
    int n  = lane & 15;
    int k0 = (lane >> 4) << 3;
    const float* p = Wt + n * ldw + k0;
    v16bf b;
#pragma unroll
    for (int i = 0; i < 8; ++i) { b[i] = (__bf16)p[i]; b[8 + i] = (__bf16)p[16 + i]; }
    return b;
}
// C/D (16x16 f32): row m = r + 8*(lane>>4), col n = lane&15
__device__ inline void store_c_lds_bf16(__bf16* C, int ldc, int lane, v8f acc) {
    int n  = lane & 15;
    int m0 = (lane >> 4) << 3;
#pragma unroll
    for (int r = 0; r < 8; ++r) C[(m0 + r) * ldc + n] = (__bf16)acc[r];
}
// Transposed C store: write D tile into Ct (N x M row-major) -> one ds_store_b128
__device__ inline void store_cT_lds_bf16(__bf16* Ct, int ldct, int lane, v8f acc) {
    int n  = lane & 15;
    int m0 = (lane >> 4) << 3;
    v8bf t;
#pragma unroll
    for (int r = 0; r < 8; ++r) t[r] = (__bf16)acc[r];
    *(v8bf*)(Ct + n * ldct + m0) = t;
}
__device__ inline void store_c_lds_f32(float* C, int ldc, int lane, v8f acc) {
    int n  = lane & 15;
    int m0 = (lane >> 4) << 3;
#pragma unroll
    for (int r = 0; r < 8; ++r) C[(m0 + r) * ldc + n] = acc[r];
}
__device__ inline v8f wmma_bf16(v16bf a, v16bf b, v8f c) {
    return __builtin_amdgcn_wmma_f32_16x16x32_bf16(false, a, false, b, (short)0, c,
                                                   false, false);
}

// ---------------- kernel 1: fused per-(n,h) row attention ---------------------
// LDS layout (dynamic, 224 KB), with phase overlays:
//   R0 [  0K, 64K)  XbT bf16[128][256] (A of q; Bt of kT/V) -> att f32[128][128]
//                   -> outf f32 rows c=0..127
//   R1 [ 64K,128K)  Vs  bf16[256][128]
//   R2 [128K,144K)  qs  bf16[128][64]   \
//   R3 [144K,160K)  kb  bf16[128][64]   / -> attb bf16[128][128]
//   R4 [160K,224K)  outf f32 rows c=128..255
//   TDM staging: Xf f32[256][128] overlays [64K,192K) (dead at load time)
__global__ __launch_bounds__(256) void attn_row_kernel(
    const float* __restrict__ x, const float* __restrict__ Wq,
    const float* __restrict__ Wk, const float* __restrict__ Wv,
    float* __restrict__ out, float* __restrict__ rowsum, float* __restrict__ rowmax)
{
    extern __shared__ char smem[];
    __bf16* XbT  = (__bf16*)(smem);
    float*  att  = (float*) (smem);
    float*  outlo= (float*) (smem);
    float*  Xf   = (float*) (smem + 64 * 1024);   // TDM staging (load phase only)
    __bf16* Vs   = (__bf16*)(smem + 64 * 1024);
    __bf16* qs   = (__bf16*)(smem + 128 * 1024);
    __bf16* kb   = (__bf16*)(smem + 144 * 1024);
    __bf16* attb = (__bf16*)(smem + 128 * 1024);
    float*  outhi= (float*) (smem + 160 * 1024);

    const int tid  = threadIdx.x;
    const int wave = tid >> 5;
    const int lane = tid & 31;
    const int nn   = blockIdx.x >> 7;   // batch
    const int hh   = blockIdx.x & 127;  // row
    const float* xrow = x + (size_t)nn * IN_C * HW + (size_t)hh * W;

    // warm L2/WGP$ for the shared weights (global_prefetch_b8)
    __builtin_prefetch(Wv + tid * 256, 0, 1);   // 256 KB, 1 KB apart
    __builtin_prefetch(Wq + tid * 64, 0, 1);    // 64 KB
    __builtin_prefetch(Wk + tid * 64, 0, 1);    // 64 KB

    // ---- TDM: DMA the (256 x 128) f32 X slab into LDS staging -----------------
    // D# per CDNA5 ISA ch.8: 2D tensor, data_size=4B, tile 128(w) x 256(c),
    // row stride = H*W elements. One tensor_load_to_lds per workgroup (wave 0).
    if (wave == 0) {
        unsigned long long ga = (unsigned long long)(uintptr_t)xrow;
        v4u g0;
        g0[0] = 1u;                                   // count=1, user mode
        g0[1] = 64u * 1024u;                          // lds_addr (bytes)
        g0[2] = (unsigned)(ga & 0xffffffffu);         // global_addr[31:0]
        g0[3] = (unsigned)((ga >> 32) & 0x01ffffffu)  // global_addr[56:32]
              | 0x80000000u;                          // type=2 ("image")
        v8i g1;
        g1[0] = (int)(2u << 16);        // data_size=2 (4 bytes); wg_mask=0
        g1[1] = (int)(128u << 16);      // tensor_dim0[15:0]=128 @bits63:48
        g1[2] = (int)(256u << 16);      // tensor_dim1[15:0]=256 @bits95:80
        g1[3] = (int)(128u << 16);      // tile_dim0=128 @bits127:112
        g1[4] = (int)256;               // tile_dim1=256, tile_dim2=0
        g1[5] = (int)HW;                // tensor_dim0_stride lo32 = 16384
        g1[6] = 0;                      // stride hi / dim1_stride lo
        g1[7] = 0;
        v4i gz; gz[0] = gz[1] = gz[2] = gz[3] = 0;    // 2D: groups 2/3 unused
#if defined(__clang_major__) && (__clang_major__ >= 23)
        v8i gz8; for (int i = 0; i < 8; ++i) gz8[i] = 0;
        __builtin_amdgcn_tensor_load_to_lds(g0, g1, gz, gz, gz8, 0);
#else
        __builtin_amdgcn_tensor_load_to_lds(g0, g1, gz, gz, 0);
#endif
        __builtin_amdgcn_s_wait_tensorcnt(0);
    }
    __syncthreads();

    // ---- convert + transpose: Xf(c,w) f32 -> XbT(w,c) bf16 (packed stores) ----
    for (int it = 0; it < 64; ++it) {
        int p  = it * 256 + tid;        // pair index over (w, c/2)
        int wi = p & 127;
        int ci = (p >> 7) << 1;
        float f0 = Xf[ci * W + wi];
        float f1 = Xf[(ci + 1) * W + wi];
        v2bf pk; pk[0] = (__bf16)f0; pk[1] = (__bf16)f1;
        *(v2bf*)(XbT + wi * IN_C + ci) = pk;
    }
    __syncthreads();

    // ---- phase A: q = X^T * Wq^T   (128x64, 8x4 tiles) ----
    for (int t = wave; t < 32; t += 8) {
        int mt = t >> 2, nt = t & 3;
        v8f acc = {};
#pragma unroll
        for (int kt = 0; kt < 8; ++kt) {
            v16bf a = load_a_lds(XbT + (mt * 16) * IN_C + kt * 32, IN_C, lane);
            v16bf b = load_bT_glob_f32(Wq + (nt * 16) * IN_C + kt * 32, IN_C, lane);
            acc = wmma_bf16(a, b, acc);
        }
        store_c_lds_bf16(qs + (mt * 16) * QK + nt * 16, QK, lane, acc);
    }
    // ---- kT tiles of  Wk * X  (64x128), stored TRANSPOSED as kb[j][d] ----
    for (int t = wave; t < 32; t += 8) {
        int mt = t >> 3, nt = t & 7;   // mt: d-block, nt: j-block
        v8f acc = {};
#pragma unroll
        for (int kt = 0; kt < 8; ++kt) {
            v16bf a = load_a_glob_f32(Wk + (mt * 16) * IN_C + kt * 32, IN_C, lane);
            v16bf b = load_b_ldsT(XbT + (nt * 16) * IN_C + kt * 32, IN_C, lane);
            acc = wmma_bf16(a, b, acc);
        }
        store_cT_lds_bf16(kb + (nt * 16) * QK + mt * 16, QK, lane, acc);
    }
    // ---- V = Wv * X   (256x128, 16x8 tiles), row-major (A of out-GEMM) ----
    for (int t = wave; t < 128; t += 8) {
        int mt = t >> 3, nt = t & 7;
        v8f acc = {};
#pragma unroll
        for (int kt = 0; kt < 8; ++kt) {
            v16bf a = load_a_glob_f32(Wv + (mt * 16) * IN_C + kt * 32, IN_C, lane);
            v16bf b = load_b_ldsT(XbT + (nt * 16) * IN_C + kt * 32, IN_C, lane);
            acc = wmma_bf16(a, b, acc);
        }
        store_c_lds_bf16(Vs + (mt * 16) * W + nt * 16, W, lane, acc);
    }
    __syncthreads();

    // ---- phase B: att = q * k^T  (128x128, K=64); Bt = kb row-major ----
    for (int t = wave; t < 64; t += 8) {
        int mt = t >> 3, nt = t & 7;
        v8f acc = {};
#pragma unroll
        for (int kt = 0; kt < 2; ++kt) {
            v16bf a = load_a_lds(qs + (mt * 16) * QK + kt * 32, QK, lane);
            v16bf b = load_b_ldsT(kb + (nt * 16) * QK + kt * 32, QK, lane);
            acc = wmma_bf16(a, b, acc);
        }
        store_c_lds_f32(att + (mt * 16) * W + nt * 16, W, lane, acc);
    }
    __syncthreads();

    // ---- phase C: row softmax; write normalized attn ROW-MAJOR bf16 ----
    if (tid < 128) {
        float* row = att + tid * W;
        float m = -3.4e38f;
        for (int j = 0; j < 128; ++j) m = fmaxf(m, row[j]);
        float s = 0.0f;
        for (int j = 0; j < 128; ++j) { float e = __expf(row[j] - m); row[j] = e; s += e; }
        float inv = 1.0f / s;
        __bf16* arow = attb + tid * W;
        for (int j = 0; j < 128; ++j) arow[j] = (__bf16)(row[j] * inv);
    }
    __syncthreads();

    // ---- phase D: out = V * att^T  (256x128, K=128); Bt = attb row-major ----
    for (int t = wave; t < 128; t += 8) {
        int mt = t >> 3, nt = t & 7;
        v8f acc = {};
#pragma unroll
        for (int kt = 0; kt < 4; ++kt) {
            v16bf a = load_a_lds(Vs + (mt * 16) * W + kt * 32, W, lane);
            v16bf b = load_b_ldsT(attb + (nt * 16) * W + kt * 32, W, lane);
            acc = wmma_bf16(a, b, acc);
        }
        float* Cb = (mt < 8) ? (outlo + (mt * 16) * W) : (outhi + ((mt - 8) * 16) * W);
        store_c_lds_f32(Cb + nt * 16, W, lane, acc);
    }
    __syncthreads();

    // ---- phase E: write out slab (float4, coalesced) + row sum/max ----
    float* orow = out + (size_t)nn * IN_C * HW + (size_t)hh * W;
    for (int it = 0; it < 32; ++it) {
        int idx = it * 256 + tid;            // quad index, 8192 total
        int ci = idx >> 5, wq = (idx & 31) << 2;
        const float* src = (ci < 128) ? (outlo + ci * W + wq)
                                      : (outhi + (ci - 128) * W + wq);
        *(v4f*)(orow + (size_t)ci * HW + wq) = *(const v4f*)src;
    }
    {
        const float* r = (tid < 128) ? (outlo + tid * W) : (outhi + (tid - 128) * W);
        float s = 0.0f, m = -3.4e38f;
        for (int j = 0; j < 128; ++j) { float v = r[j]; s += v; m = fmaxf(m, v); }
        size_t o = ((size_t)nn * H + hh) * IN_C + tid;
        rowsum[o] = s;
        rowmax[o] = m;
    }
}

// ---------------- kernel 2: SE gate MLP per batch (deterministic reduce) ------
__global__ __launch_bounds__(256) void gate_kernel(
    const float* __restrict__ rowsum, const float* __restrict__ rowmax,
    const float* __restrict__ W1, const float* __restrict__ W2,
    float* __restrict__ gate)
{
    __shared__ float avg[IN_C], mx[IN_C], hsum[16];
    int n = blockIdx.x, tid = threadIdx.x;
    {
        float s = 0.0f, m = -3.4e38f;
        const float* ps = rowsum + (size_t)n * H * IN_C + tid;
        const float* pm = rowmax + (size_t)n * H * IN_C + tid;
        for (int h = 0; h < H; ++h) {   // fixed order -> deterministic
            s += ps[(size_t)h * IN_C];
            m = fmaxf(m, pm[(size_t)h * IN_C]);
        }
        avg[tid] = s * (1.0f / (float)HW);
        mx[tid]  = m;
    }
    __syncthreads();
    if (tid < 16) {
        float a = 0.0f, b = 0.0f;
        for (int c = 0; c < IN_C; ++c) {
            float w = W1[tid * IN_C + c];
            a += w * avg[c];
            b += w * mx[c];
        }
        hsum[tid] = fmaxf(a, 0.0f) + fmaxf(b, 0.0f);   // relu(avg path)+relu(max path)
    }
    __syncthreads();
    float g = 0.0f;
    for (int j = 0; j < 16; ++j) g += W2[tid * 16 + j] * hsum[j];
    gate[n * IN_C + tid] = 1.0f / (1.0f + __expf(-g));
}

// ---------------- kernel 3: y = gama * gate * out + x -------------------------
__global__ __launch_bounds__(256) void finalize_kernel(
    const float* __restrict__ x, const float* __restrict__ gate,
    const float* __restrict__ gama, float* __restrict__ out)
{
    size_t idx = (size_t)blockIdx.x * 256 + threadIdx.x;
    int nc = (int)(idx >> 14);                // (n*IN_C + c), since HW = 2^14
    float g = gama[0] * gate[nc];
    out[idx] = g * out[idx] + x[idx];
}

// ---------------- launcher ----------------------------------------------------
extern "C" void kernel_launch(void* const* d_in, const int* in_sizes, int n_in,
                              void* d_out, int out_size, void* d_ws, size_t ws_size,
                              hipStream_t stream) {
    const float* x    = (const float*)d_in[0];
    const float* Wq   = (const float*)d_in[1];
    const float* Wk   = (const float*)d_in[2];
    const float* Wv   = (const float*)d_in[3];
    const float* W1   = (const float*)d_in[4];
    const float* W2   = (const float*)d_in[5];
    const float* gama = (const float*)d_in[6];
    float* out = (float*)d_out;

    // workspace: per-(n,h,c) partial reductions (1 MB each) + gate (8 KB)
    float* rowsum = (float*)d_ws;
    float* rowmax = (float*)((char*)d_ws + (size_t)NBATCH * H * IN_C * 4);
    float* gate   = (float*)((char*)d_ws + (size_t)2 * NBATCH * H * IN_C * 4);

    attn_row_kernel<<<NBATCH * H, 256, 224 * 1024, stream>>>(x, Wq, Wk, Wv, out,
                                                             rowsum, rowmax);
    gate_kernel<<<NBATCH, 256, 0, stream>>>(rowsum, rowmax, W1, W2, gate);
    finalize_kernel<<<(NBATCH * IN_C * HW) / 256, 256, 0, stream>>>(x, gate, gama, out);
}
